// Decoder_44547400794517
// MI455X (gfx1250) — matrix operand order repaired and checked by reading, not compiled, not run on hardware
//
#include <hip/hip_runtime.h>

// ---------------------------------------------------------------------------
// CDNA5 (gfx1250) attention-decoder step.  wave32, WMMA bf16 (f32 accum).
// B=64, S=128, H=1024, VOCAB=32000.
// ---------------------------------------------------------------------------

typedef __attribute__((ext_vector_type(16))) __bf16 v16bf;
typedef __attribute__((ext_vector_type(8)))  float  v8f;

#define BQ 64          // batch
#define HH 1024        // hidden
#define SS 128         // src len
#define VOC 32000

__device__ inline v8f wmma_bf16(v16bf a, v16bf b, v8f c) {
  // D = A(16x32 bf16) * B(32x16 bf16) + C(16x16 f32)
  return __builtin_amdgcn_wmma_f32_16x16x32_bf16(
      /*neg_a=*/false, a, /*neg_b=*/false, b,
      /*c_mod=*/(short)0, c, /*reuse_a=*/false, /*reuse_b=*/false);
}

// A fragment: 16x32 tile of row-major A (row stride ld) at (m0,k0).
// ISA 7.12.2 16-bit A 16x32: lane%16 = M row; lanes 0-15 hold K {0..7,16..23},
// lanes 16-31 hold K {8..15, 24..31}.  Per-lane reads are two contiguous
// 32-byte runs -> global_load_b128 pairs + v_cvt_pk_bf16_f32.
__device__ inline v16bf load_a_frag(const float* __restrict__ A, int ld,
                                    int m0, int k0) {
  const int lane = threadIdx.x & 31;
  const float* row = A + (long)(m0 + (lane & 15)) * ld
                       + k0 + ((lane & 16) >> 1);   // +8 for upper half-wave
  v16bf a;
#pragma unroll
  for (int e = 0; e < 16; ++e) a[e] = (__bf16)row[e + (e & 8)];
  return a;
}

// B fragment: 32x16 tile of row-major B (K x N, row stride ld) at (k0,n0).
// lane%16 = N col; lanes 0-15 hold K {0..7,16..23}, lanes 16-31 hold K {8..15,24..31}.
__device__ inline v16bf load_b_frag(const float* __restrict__ B, int ld,
                                    int k0, int n0) {
  const int lane = threadIdx.x & 31;
  const float* col = B + (long)(k0 + ((lane & 16) >> 1)) * ld + n0 + (lane & 15);
  v16bf b;
#pragma unroll
  for (int e = 0; e < 16; ++e) b[e] = (__bf16)col[(long)(e + (e & 8)) * ld];
  return b;
}

__device__ inline float sigmoidf_(float x) { return 1.0f / (1.0f + __expf(-x)); }

// ---------------------------------------------------------------------------
// 0) embedding gather: ex[m, :] = emb[x[m], :]
// ---------------------------------------------------------------------------
__global__ __launch_bounds__(256) void gather_emb_kernel(
    const int* __restrict__ x, const float* __restrict__ emb,
    float* __restrict__ ex) {
  int i = blockIdx.x * 256 + threadIdx.x;          // 64*1024 threads
  int m = i >> 10, e = i & (HH - 1);
  ex[i] = emb[(long)x[m] * HH + e];
}

// ---------------------------------------------------------------------------
// 1) weight-bound M=64 GEMM:  C = A(64xK) @ B(KxN) + bias.
//    Register-blocked over M: one wave owns a full 64-row x 16-col stripe,
//    so each weight (B) fragment is loaded ONCE and feeds 4 WMMAs.
//    Waves needed = N/16;  8 waves per workgroup.
// ---------------------------------------------------------------------------
__global__ __launch_bounds__(256) void gemm_bias_kernel(
    const float* __restrict__ A, const float* __restrict__ B,
    const float* __restrict__ bias, float* __restrict__ C, int K, int N) {
  const int w  = blockIdx.x * 8 + (threadIdx.x >> 5);
  const int n0 = w << 4;
  if (n0 >= N) return;                       // wave-uniform
  const int lane = threadIdx.x & 31;

  v8f acc0 = {}, acc1 = {}, acc2 = {}, acc3 = {};
  for (int k0 = 0; k0 < K; k0 += 32) {
    // prefetch next K-block of the weight stripe (global_prefetch_b8)
    if (k0 + 32 < K)
      __builtin_prefetch(B + (long)(k0 + 32 + (lane & 16)) * N + n0 + (lane & 15),
                         0, 1);
    v16bf b  = load_b_frag(B, N, k0, n0);
    v16bf a0 = load_a_frag(A, K,  0, k0);
    v16bf a1 = load_a_frag(A, K, 16, k0);
    v16bf a2 = load_a_frag(A, K, 32, k0);
    v16bf a3 = load_a_frag(A, K, 48, k0);
    acc0 = wmma_bf16(a0, b, acc0);
    acc1 = wmma_bf16(a1, b, acc1);
    acc2 = wmma_bf16(a2, b, acc2);
    acc3 = wmma_bf16(a3, b, acc3);
  }
  const int n  = n0 + (lane & 15);
  const int mo = (lane >> 4) << 3;
  const float bv = bias[n];
#pragma unroll
  for (int r = 0; r < 8; ++r) {
    C[(long)(r + mo)      * N + n] = acc0[r] + bv;
    C[(long)(r + mo + 16) * N + n] = acc1[r] + bv;
    C[(long)(r + mo + 32) * N + n] = acc2[r] + bv;
    C[(long)(r + mo + 48) * N + n] = acc3[r] + bv;
  }
}

// ---------------------------------------------------------------------------
// 2) attention score, fully fused:
//    score[b,s] = sum_n tanh( (enc@W2)[b,s,n] + q[b,n] + b2[n] ) * Vw[n]
//    One workgroup per 16 flattened (b,s) rows (rows share b since S=128).
//    The (B,S,H) tanh intermediate is never materialized.
// ---------------------------------------------------------------------------
__global__ __launch_bounds__(256) void attn_score_kernel(
    const float* __restrict__ enc, const float* __restrict__ W2,
    const float* __restrict__ b2,  const float* __restrict__ q,
    const float* __restrict__ Vw,  float* __restrict__ score) {
  const int row0 = blockIdx.x * 16;          // 512 blocks, flattened b*128+s
  const int b    = row0 >> 7;
  const int wave = threadIdx.x >> 5;
  const int lane = threadIdx.x & 31;

  __shared__ float s_score[16];
  if (threadIdx.x < 16) s_score[threadIdx.x] = 0.0f;
  __syncthreads();

  float sacc[8] = {0.f, 0.f, 0.f, 0.f, 0.f, 0.f, 0.f, 0.f};

  for (int pass = 0; pass < 8; ++pass) {     // 8 waves x 8 passes = 64 n-tiles
    const int n0 = ((pass << 3) + wave) << 4;
    v8f acc = {};
    for (int k0 = 0; k0 < HH; k0 += 32) {
      v16bf a = load_a_frag(enc, HH, row0, k0);   // enc viewed as (8192,1024)
      v16bf w = load_b_frag(W2, HH, k0, n0);
      acc = wmma_bf16(a, w, acc);
    }
    const int n  = n0 + (lane & 15);
    const float qv = q[b * HH + n] + b2[n];
    const float vw = Vw[n];
#pragma unroll
    for (int r = 0; r < 8; ++r) sacc[r] += tanhf(acc[r] + qv) * vw;
  }

  // reduce over the 16 n-lanes (xor 1,2,4,8 stays inside each 16-lane half)
#pragma unroll
  for (int r = 0; r < 8; ++r) {
    float v = sacc[r];
    v += __shfl_xor(v, 1);
    v += __shfl_xor(v, 2);
    v += __shfl_xor(v, 4);
    v += __shfl_xor(v, 8);
    if ((lane & 15) == 0)
      atomicAdd(&s_score[r + ((lane >> 4) << 3)], v);   // ds_add_f32
  }
  __syncthreads();
  if (threadIdx.x < 16) score[row0 + threadIdx.x] = s_score[threadIdx.x];
}

// ---------------------------------------------------------------------------
// 3) softmax over S + context = sum_s aw[s]*enc[b,s,:].  One WG per batch row.
//    (Vb dropped: softmax is shift-invariant.)
// ---------------------------------------------------------------------------
__global__ __launch_bounds__(256) void softmax_context_kernel(
    const float* __restrict__ score, const float* __restrict__ enc,
    float* __restrict__ aw_out, float* __restrict__ ctx) {
  const int b = blockIdx.x, tid = threadIdx.x;
  __shared__ float s_aw[SS];
  __shared__ float s_red[64];

  if (tid < SS) s_aw[tid] = score[b * SS + tid];
  __syncthreads();
  if (tid < 64) s_red[tid] = fmaxf(s_aw[tid], s_aw[tid + 64]);
  __syncthreads();
  for (int off = 32; off > 0; off >>= 1) {
    if (tid < off) s_red[tid] = fmaxf(s_red[tid], s_red[tid + off]);
    __syncthreads();
  }
  const float mx = s_red[0];
  __syncthreads();
  if (tid < SS) s_aw[tid] = __expf(s_aw[tid] - mx);
  __syncthreads();
  if (tid < 64) s_red[tid] = s_aw[tid] + s_aw[tid + 64];
  __syncthreads();
  for (int off = 32; off > 0; off >>= 1) {
    if (tid < off) s_red[tid] = s_red[tid] + s_red[tid + off];
    __syncthreads();
  }
  const float inv = 1.0f / s_red[0];
  __syncthreads();
  if (tid < SS) {
    s_aw[tid] *= inv;
    aw_out[b * SS + tid] = s_aw[tid];
  }
  __syncthreads();
  for (int h = tid; h < HH; h += 256) {
    float acc = 0.0f;
    for (int s = 0; s < SS; ++s)
      acc += s_aw[s] * enc[((long)b * SS + s) * HH + h];
    ctx[b * HH + h] = acc;
  }
}

// ---------------------------------------------------------------------------
// 4) LSTM layer (h0 == 0  =>  Wr term is zero and never read;
//    f-gate multiplied by 0 => its Wk columns never read).
//    g = [ctx | hprev] @ Wk[:, {i,gc,o}] + bl ;  h = sig(o)*tanh(sig(i)*tanh(gc))
//    Each wave: one 16x16 tile per gate, shared A fragment, 3 WMMAs / K-step.
//    The 4 row-tile waves sharing a weight tile are in the same workgroup
//    (WGP$ reuse).
// ---------------------------------------------------------------------------
__global__ __launch_bounds__(256) void lstm_layer_kernel(
    const float* __restrict__ ctx, const float* __restrict__ hprev,
    const float* __restrict__ Wk,  const float* __restrict__ bl,
    float* __restrict__ hout, float* __restrict__ hout2) {
  const int w  = blockIdx.x * 8 + (threadIdx.x >> 5);
  const int m0 = (w & 3) << 4;
  const int n0 = (w >> 2) << 4;                    // 64 col tiles
  v8f acc_i = {}, acc_g = {}, acc_o = {};
  for (int k0 = 0; k0 < 2 * HH; k0 += 32) {
    const float* Asrc = (k0 < HH) ? ctx : hprev;   // concat without a copy
    const int    ka   = (k0 < HH) ? k0 : (k0 - HH);
    v16bf a  = load_a_frag(Asrc, HH, m0, ka);
    v16bf bi = load_b_frag(Wk, 4 * HH, k0, n0);             // i gate
    v16bf bg = load_b_frag(Wk, 4 * HH, k0, n0 + 2 * HH);    // gc gate
    v16bf bo = load_b_frag(Wk, 4 * HH, k0, n0 + 3 * HH);    // o gate
    acc_i = wmma_bf16(a, bi, acc_i);
    acc_g = wmma_bf16(a, bg, acc_g);
    acc_o = wmma_bf16(a, bo, acc_o);
  }
  const int lane = threadIdx.x & 31;
  const int n = n0 + (lane & 15);
  const float bi_ = bl[n], bg_ = bl[n + 2 * HH], bo_ = bl[n + 3 * HH];
#pragma unroll
  for (int r = 0; r < 8; ++r) {
    const int m = m0 + r + ((lane >> 4) << 3);
    float iv = sigmoidf_(acc_i[r] + bi_);
    float gv = tanhf(acc_g[r] + bg_);
    float ov = sigmoidf_(acc_o[r] + bo_);
    float hv = ov * tanhf(iv * gv);
    hout[(long)m * HH + n] = hv;
    if (hout2) hout2[(long)m * HH + n] = hv;
  }
}

// ---------------------------------------------------------------------------
// launch
// ---------------------------------------------------------------------------
extern "C" void kernel_launch(void* const* d_in, const int* in_sizes, int n_in,
                              void* d_out, int out_size, void* d_ws,
                              size_t ws_size, hipStream_t stream) {
  (void)in_sizes; (void)n_in; (void)out_size; (void)ws_size;

  const int*   x      = (const int*)  d_in[0];
  const float* hidden = (const float*)d_in[1];
  const float* enc    = (const float*)d_in[2];
  const float* W1     = (const float*)d_in[3];
  const float* b1     = (const float*)d_in[4];
  const float* W2     = (const float*)d_in[5];
  const float* b2     = (const float*)d_in[6];
  const float* Vw     = (const float*)d_in[7];
  /* d_in[8] = Vb: softmax shift-invariant, unused */
  const float* emb    = (const float*)d_in[9];
  const float* Wk[4]  = {(const float*)d_in[10], (const float*)d_in[13],
                         (const float*)d_in[16], (const float*)d_in[19]};
  /* d_in[11,14,17,20] = Wr1..4: multiplied by h0==0, never read */
  const float* bl[4]  = {(const float*)d_in[12], (const float*)d_in[15],
                         (const float*)d_in[18], (const float*)d_in[21]};
  const float* fcW    = (const float*)d_in[22];
  const float* fcb    = (const float*)d_in[23];

  float* out    = (float*)d_out;
  float* logits = out;                                   // 64*32000
  float* h4_out = out + (long)BQ * VOC;                  // 64*1024
  float* aw_out = out + (long)BQ * VOC + BQ * HH;        // 64*128

  float* ws    = (float*)d_ws;
  float* q     = ws;                          // 65536
  float* score = ws + 65536;                  // 8192
  float* ctx   = ws + 73728;                  // 65536
  float* hbuf[5] = {ws + 139264, ws + 204800, ws + 270336,
                    ws + 335872, ws + 401408};           // ex,h1..h4 (64K ea)

  // 0) ex = emb[x]
  gather_emb_kernel<<<BQ * HH / 256, 256, 0, stream>>>(x, emb, hbuf[0]);
  // 1) q = hidden @ W1 + b1          (waves = 1024/16 = 64 -> 8 WGs)
  gemm_bias_kernel<<<8, 256, 0, stream>>>(hidden, W1, b1, q, HH, HH);
  // 2) score (fused enc@W2 + tanh + .Vw reduction)
  attn_score_kernel<<<BQ * SS / 16, 256, 0, stream>>>(enc, W2, b2, q, Vw, score);
  // 3) softmax + context (+ write attention_weights output)
  softmax_context_kernel<<<BQ, 256, 0, stream>>>(score, enc, aw_out, ctx);
  // 4) LSTM stack (layer 4 also writes the h4 output region)
  for (int l = 0; l < 4; ++l)
    lstm_layer_kernel<<<32, 256, 0, stream>>>(ctx, hbuf[l], Wk[l], bl[l],
                                              hbuf[l + 1],
                                              (l == 3) ? h4_out : nullptr);
  // 5) logits = h4 @ fcW + fcb      (waves = 32000/16 = 2000 -> 250 WGs)
  gemm_bias_kernel<<<250, 256, 0, stream>>>(hbuf[4], fcW, fcb, logits,
                                            HH, VOC);
}
